// SpatialSpectralAttention_25958782337667
// MI455X (gfx1250) — matrix operand order repaired and checked by reading, not compiled, not run on hardware
//
#include <hip/hip_runtime.h>
#include <hip/hip_bf16.h>

typedef __bf16 bf16_t;
typedef __attribute__((ext_vector_type(8)))  bf16_t v8bf;
typedef __attribute__((ext_vector_type(16))) bf16_t v16bf;
typedef __attribute__((ext_vector_type(8)))  float  v8f;

union ABFrag { v16bf v; v8bf h[2]; };

#define WMMA_BF16(A,B,C) __builtin_amdgcn_wmma_f32_16x16x32_bf16(false,(A),false,(B),(short)0,(C),false,false)

static __device__ __forceinline__ v8f zero8() {
    v8f z = {0.f,0.f,0.f,0.f,0.f,0.f,0.f,0.f};
    return z;
}

#define SCALE_F 0.0883883476483184406f  /* (2*64)^-0.5 */

// ---------------------------------------------------------------------------
// Prep kernels: build bf16 concatenated inputs / combined weights
// ---------------------------------------------------------------------------

// Xc[8192][1536] = [bf16(rgb) | bf16(spec)]
__global__ void prep_x(const float* __restrict__ rgb, const float* __restrict__ spec,
                       bf16_t* __restrict__ Xc) {
    int i = blockIdx.x * 256 + threadIdx.x;          // exactly 8192*768 threads
    int row = i / 768, col = i - row * 768;
    Xc[(size_t)row * 1536 + col]       = (bf16_t)rgb[i];
    Xc[(size_t)row * 1536 + 768 + col] = (bf16_t)spec[i];
}

// Wc[2304][1536] = [s_r*W_rgb | s_s*W_spec]; bc[2304] combined bias; PW bf16 proj_w
__global__ void prep_w(const float* __restrict__ rw, const float* __restrict__ rb,
                       const float* __restrict__ sw, const float* __restrict__ sb,
                       const float* __restrict__ pw,
                       bf16_t* __restrict__ Wc, float* __restrict__ bc,
                       bf16_t* __restrict__ PW) {
    int i = blockIdx.x * 256 + threadIdx.x;          // covers 2304*768
    if (i < 2304 * 768) {
        int r = i / 768, col = i - r * 768;
        int t = r / 768;                              // 0=q 1=k 2=v
        float sr = (t == 0) ? SCALE_F : 1.0f;
        float ss = (t == 0) ? SCALE_F : ((t == 1) ? 1.0f : 0.5f);
        Wc[(size_t)r * 1536 + col]       = (bf16_t)(rw[i] * sr);
        Wc[(size_t)r * 1536 + 768 + col] = (bf16_t)(sw[i] * ss);
    }
    if (i < 768 * 768) PW[i] = (bf16_t)pw[i];
    if (i < 2304) {
        int t = i / 768;
        float v = (t == 0) ? SCALE_F * (rb[i] + sb[i])
                : (t == 1) ? (rb[i] + sb[i])
                           : (rb[i] + 0.5f * sb[i]);
        bc[i] = v;
    }
}

// ---------------------------------------------------------------------------
// QKV GEMM: [8192,1536] x [2304,1536]^T -> Q,K [B,H,L,64], V^T [B,H,64,L] (bf16)
// One wave computes a 32x64 output tile; 48 K-steps of 32.
// ---------------------------------------------------------------------------
__global__ void qkv_gemm(const bf16_t* __restrict__ Xc, const bf16_t* __restrict__ Wc,
                         const float* __restrict__ bc,
                         bf16_t* __restrict__ Qg, bf16_t* __restrict__ Kg,
                         bf16_t* __restrict__ VTg) {
    int tid = threadIdx.x, w = tid >> 5, lane = tid & 31;
    int half = lane >> 4, lr = lane & 15;
    int W = blockIdx.x * 8 + w;                      // 9216 waves
    int tm = W / 36, tn = W - tm * 36;
    int r0 = tm * 32, c0 = tn * 64;

    v8f acc[2][4];
#pragma unroll
    for (int i = 0; i < 2; ++i)
#pragma unroll
        for (int j = 0; j < 4; ++j) acc[i][j] = zero8();

    for (int kt = 0; kt < 48; ++kt) {
        ABFrag a[2];
#pragma unroll
        for (int i = 0; i < 2; ++i) {
            const bf16_t* ab = Xc + (size_t)(r0 + 16 * i + lr) * 1536 + kt * 32;
            a[i].h[0] = *(const v8bf*)(ab + 8 * half);
            a[i].h[1] = *(const v8bf*)(ab + 16 + 8 * half);
        }
#pragma unroll
        for (int j = 0; j < 4; ++j) {
            ABFrag bfr;
            const bf16_t* bb = Wc + (size_t)(c0 + 16 * j + lr) * 1536 + kt * 32;
            bfr.h[0] = *(const v8bf*)(bb + 8 * half);
            bfr.h[1] = *(const v8bf*)(bb + 16 + 8 * half);
#pragma unroll
            for (int i = 0; i < 2; ++i)
                acc[i][j] = WMMA_BF16(a[i].v, bfr.v, acc[i][j]);
        }
    }

#pragma unroll
    for (int i = 0; i < 2; ++i)
#pragma unroll
        for (int j = 0; j < 4; ++j) {
            int c = c0 + 16 * j + lr;
            float bias = bc[c];
            int t = c / 768, cc = c - t * 768;
            int hh = cc >> 6, hd = cc & 63;
#pragma unroll
            for (int r = 0; r < 8; ++r) {
                int row = r0 + 16 * i + r + 8 * half;
                int b = row >> 10, l = row & 1023;
                float v = acc[i][j][r] + bias;
                if (t == 0)
                    Qg[((size_t)(b * 12 + hh) * 1024 + l) * 64 + hd] = (bf16_t)v;
                else if (t == 1)
                    Kg[((size_t)(b * 12 + hh) * 1024 + l) * 64 + hd] = (bf16_t)v;
                else
                    VTg[((size_t)(b * 12 + hh) * 64 + hd) * 1024 + l] = (bf16_t)v;
            }
        }
}

// ---------------------------------------------------------------------------
// Flash attention: block = 8 waves on one (b,h); each wave owns a 16-row Q
// tile; K (32x64) and V^T (64x32) tiles staged in LDS and shared by all waves.
// ---------------------------------------------------------------------------
__global__ void attn(const bf16_t* __restrict__ Qg, const bf16_t* __restrict__ Kg,
                     const bf16_t* __restrict__ VTg, bf16_t* __restrict__ AO) {
    __shared__ bf16_t ldsK[32 * 72];       // padded rows: conflict-free, 16B aligned
    __shared__ bf16_t ldsV[64 * 40];
    __shared__ bf16_t ldsP[8][16 * 40];    // per-wave P relayout scratch

    int tid = threadIdx.x, w = tid >> 5, lane = tid & 31;
    int half = lane >> 4, lr = lane & 15;
    int bh = blockIdx.x >> 3;
    int b = bh / 12, hh = bh - b * 12;
    int qbase = (blockIdx.x & 7) * 128 + w * 16;

    // Q A-fragments (two 32-wide hd slices), loaded once
    ABFrag qa[2];
    const bf16_t* qrow = Qg + ((size_t)bh * 1024 + qbase + lr) * 64;
#pragma unroll
    for (int s = 0; s < 2; ++s) {
        qa[s].h[0] = *(const v8bf*)(qrow + s * 32 + 8 * half);
        qa[s].h[1] = *(const v8bf*)(qrow + s * 32 + 16 + 8 * half);
    }

    v8f o[4];
#pragma unroll
    for (int j = 0; j < 4; ++j) o[j] = zero8();
    float mprev[8], lrun[8];
#pragma unroll
    for (int r = 0; r < 8; ++r) { mprev[r] = -INFINITY; lrun[r] = 0.f; }

    int krow = tid >> 3, kc = tid & 7;     // K tile: 32 rows x 8 chunks of 16B
    int vrow = tid >> 2, vc = tid & 3;     // VT tile: 64 rows x 4 chunks of 16B
    const bf16_t* Kbh = Kg + (size_t)bh * 1024 * 64;
    const bf16_t* Vbh = VTg + (size_t)bh * 64 * 1024;

    for (int it = 0; it < 64; ++it) {
        int kv0 = it * 32;
        v8bf kd = *(const v8bf*)(Kbh + (size_t)(kv0 + krow) * 64 + kc * 8);
        v8bf vd = *(const v8bf*)(Vbh + (size_t)vrow * 1024 + kv0 + vc * 8);
        __syncthreads();                               // prev iter done reading
        *(v8bf*)&ldsK[krow * 72 + kc * 8] = kd;
        *(v8bf*)&ldsV[vrow * 40 + vc * 8] = vd;
        __syncthreads();                               // tiles visible

        // S = Q K^T  (16x32 scores: two 16x16 accumulators)
        v8f s0 = zero8(), s1 = zero8();
        ABFrag kb;
#pragma unroll
        for (int nb = 0; nb < 2; ++nb) {
#pragma unroll
            for (int s = 0; s < 2; ++s) {
                const bf16_t* kp = &ldsK[(nb * 16 + lr) * 72 + s * 32];
                kb.h[0] = *(const v8bf*)(kp + 8 * half);
                kb.h[1] = *(const v8bf*)(kp + 16 + 8 * half);
                if (nb == 0) s0 = WMMA_BF16(qa[s].v, kb.v, s0);
                else         s1 = WMMA_BF16(qa[s].v, kb.v, s1);
            }
        }

        // online softmax (row r lives in vgpr r; 16 lanes of each half = columns)
#pragma unroll
        for (int r = 0; r < 8; ++r) {
            float t = fmaxf(s0[r], s1[r]);
            t = fmaxf(t, __shfl_xor(t, 1));
            t = fmaxf(t, __shfl_xor(t, 2));
            t = fmaxf(t, __shfl_xor(t, 4));
            t = fmaxf(t, __shfl_xor(t, 8));
            float mn = fmaxf(mprev[r], t);
            float al = __expf(mprev[r] - mn);
            float e0 = __expf(s0[r] - mn);
            float e1 = __expf(s1[r] - mn);
            float rs = e0 + e1;
            rs += __shfl_xor(rs, 1);
            rs += __shfl_xor(rs, 2);
            rs += __shfl_xor(rs, 4);
            rs += __shfl_xor(rs, 8);
            lrun[r] = lrun[r] * al + rs;
            mprev[r] = mn;
#pragma unroll
            for (int j = 0; j < 4; ++j) o[j][r] *= al;
            int rr = r + 8 * half;                      // relayout P via LDS
            ldsP[w][rr * 40 + lr]      = (bf16_t)e0;
            ldsP[w][rr * 40 + 16 + lr] = (bf16_t)e1;
        }
        asm volatile("s_wait_dscnt 0" ::: "memory");

        // P A-fragment
        ABFrag pf;
        pf.h[0] = *(const v8bf*)&ldsP[w][lr * 40 + 8 * half];
        pf.h[1] = *(const v8bf*)&ldsP[w][lr * 40 + 16 + 8 * half];

        // O += P V   (V^T tile: lane n holds V[8h..8h+7 and 16+8h..16+8h+7, 16j+n])
#pragma unroll
        for (int j = 0; j < 4; ++j) {
            ABFrag vb;
            const bf16_t* vp = &ldsV[(16 * j + lr) * 40];
            vb.h[0] = *(const v8bf*)(vp + 8 * half);
            vb.h[1] = *(const v8bf*)(vp + 16 + 8 * half);
            o[j] = WMMA_BF16(pf.v, vb.v, o[j]);
        }
    }

    // normalize + store attention output (bf16, [B,L,D] with col = h*64+hd)
#pragma unroll
    for (int r = 0; r < 8; ++r) {
        float inv = 1.0f / lrun[r];
        int seq = qbase + r + 8 * half;
        size_t rowb = ((size_t)b * 1024 + seq) * 768 + hh * 64;
#pragma unroll
        for (int j = 0; j < 4; ++j)
            AO[rowb + j * 16 + lr] = (bf16_t)(o[j][r] * inv);
    }
}

// ---------------------------------------------------------------------------
// Output projection: [8192,768](bf16) x [768,768]^T(bf16) + bias -> f32
// ---------------------------------------------------------------------------
__global__ void proj_gemm(const bf16_t* __restrict__ AO, const bf16_t* __restrict__ PW,
                          const float* __restrict__ pb, float* __restrict__ out) {
    int tid = threadIdx.x, w = tid >> 5, lane = tid & 31;
    int half = lane >> 4, lr = lane & 15;
    int W = blockIdx.x * 8 + w;                       // 3072 waves
    int tm = W / 12, tn = W - tm * 12;
    int r0 = tm * 32, c0 = tn * 64;

    v8f acc[2][4];
#pragma unroll
    for (int i = 0; i < 2; ++i)
#pragma unroll
        for (int j = 0; j < 4; ++j) acc[i][j] = zero8();

    for (int kt = 0; kt < 24; ++kt) {
        ABFrag a[2];
#pragma unroll
        for (int i = 0; i < 2; ++i) {
            const bf16_t* ab = AO + (size_t)(r0 + 16 * i + lr) * 768 + kt * 32;
            a[i].h[0] = *(const v8bf*)(ab + 8 * half);
            a[i].h[1] = *(const v8bf*)(ab + 16 + 8 * half);
        }
#pragma unroll
        for (int j = 0; j < 4; ++j) {
            ABFrag bfr;
            const bf16_t* bb = PW + (size_t)(c0 + 16 * j + lr) * 768 + kt * 32;
            bfr.h[0] = *(const v8bf*)(bb + 8 * half);
            bfr.h[1] = *(const v8bf*)(bb + 16 + 8 * half);
#pragma unroll
            for (int i = 0; i < 2; ++i)
                acc[i][j] = WMMA_BF16(a[i].v, bfr.v, acc[i][j]);
        }
    }

#pragma unroll
    for (int i = 0; i < 2; ++i)
#pragma unroll
        for (int j = 0; j < 4; ++j) {
            int c = c0 + 16 * j + lr;
            float bias = pb[c];
#pragma unroll
            for (int r = 0; r < 8; ++r) {
                int row = r0 + 16 * i + r + 8 * half;
                out[(size_t)row * 768 + c] = acc[i][j][r] + bias;
            }
        }
}

// ---------------------------------------------------------------------------
extern "C" void kernel_launch(void* const* d_in, const int* in_sizes, int n_in,
                              void* d_out, int out_size, void* d_ws, size_t ws_size,
                              hipStream_t stream) {
    const float* rgb   = (const float*)d_in[0];
    const float* spec  = (const float*)d_in[1];
    const float* rqkvw = (const float*)d_in[2];
    const float* rqkvb = (const float*)d_in[3];
    const float* sqkvw = (const float*)d_in[4];
    const float* sqkvb = (const float*)d_in[5];
    const float* pw    = (const float*)d_in[6];
    const float* pb    = (const float*)d_in[7];
    float* out = (float*)d_out;

    char* ws = (char*)d_ws;
    bf16_t* Xc  = (bf16_t*)ws; ws += (size_t)8192 * 1536 * 2;   // 25.2 MB
    bf16_t* Wc  = (bf16_t*)ws; ws += (size_t)2304 * 1536 * 2;   //  7.1 MB
    float*  bc  = (float* )ws; ws += (size_t)2304 * 4;          //  9 KB
    bf16_t* Qg  = (bf16_t*)ws; ws += (size_t)96 * 1024 * 64 * 2; // 12.6 MB
    bf16_t* Kg  = (bf16_t*)ws; ws += (size_t)96 * 1024 * 64 * 2;
    bf16_t* VTg = (bf16_t*)ws; ws += (size_t)96 * 64 * 1024 * 2;
    bf16_t* AO  = (bf16_t*)ws; ws += (size_t)8192 * 768 * 2;     // 12.6 MB
    bf16_t* PW  = (bf16_t*)ws; ws += (size_t)768 * 768 * 2;      //  1.2 MB

    prep_x   <<<24576, 256, 0, stream>>>(rgb, spec, Xc);
    prep_w   <<< 6912, 256, 0, stream>>>(rqkvw, rqkvb, sqkvw, sqkvb, pw, Wc, bc, PW);
    qkv_gemm <<< 1152, 256, 0, stream>>>(Xc, Wc, bc, Qg, Kg, VTg);
    attn     <<<  768, 256, 0, stream>>>(Qg, Kg, VTg, AO);
    proj_gemm<<<  384, 256, 0, stream>>>(AO, PW, pb, out);
}
